// Attention_57337813401944
// MI455X (gfx1250) — matrix operand order repaired and checked by reading, not compiled, not run on hardware
//
#include <hip/hip_runtime.h>
#include <hip/hip_bf16.h>

typedef __attribute__((ext_vector_type(16))) __bf16 v16bf;
typedef __attribute__((ext_vector_type(8)))  float  v8f;

#define B_SZ   2
#define N_SEQ  2048
#define DIMD   1024
#define NH     16
#define DHD    64
#define QKV_COLS (3 * NH * DHD)   // 3072
#define M_ROWS   (B_SZ * N_SEQ)   // 4096
#define HEAD_ELEMS ((long)N_SEQ * DHD)              // 131072 per (b,h)
#define TENSOR_ELEMS ((long)B_SZ * NH * HEAD_ELEMS) // 4,194,304 per tensor

// ---------------------------------------------------------------------------
// Fragment-linear layouts (CDNA5 ISA 7.12.2, wave32, 16x16x32 bf16 WMMA):
//  A frag  (16x32): lane L = m + 16*((k>>3)&1), elem e = (k>>4)*8 + (k&7)
//  B frag  (32x16): lane L = n + 16*((k>>3)&1), elem e = (k>>4)*8 + (k&7)
//  C/D     (16x16): VGPR j -> M = j + 8*(lane>>4), N = lane&15
// A fragment-linear tile stores lane L's 16 bf16 elements contiguously at
// offset L*16: every fragment load is one 32-byte (2 x ds/global b128) read.
// Each 32-lane fragment = 512 bf16 elements (1 KB).
// ---------------------------------------------------------------------------

static __device__ inline unsigned pack2bf(float a, float b) {
  union { __bf16 h[2]; unsigned u; } cv;
  cv.h[0] = (__bf16)a;
  cv.h[1] = (__bf16)b;
  return cv.u;
}

static __device__ inline v16bf frag_ld(const __bf16* base) {
  return *((const v16bf*)base + (threadIdx.x & 31));
}

// Single-instruction xor-shuffle within 16-lane groups (ds_swizzle, no mem).
template <int MASK>
static __device__ inline float lane_xor_f(float v) {
  int i = __builtin_bit_cast(int, v);
  i = __builtin_amdgcn_ds_swizzle(i, (MASK << 10) | 0x1f); // xor=MASK, and=0x1f
  return __builtin_bit_cast(float, i);
}

// CDNA5 async global->LDS copy (ASYNCcnt path), 16B per lane per issue.
static __device__ inline void async_copy_b128(unsigned lds_off, const void* gptr) {
  unsigned long long ga = (unsigned long long)(size_t)gptr;
  asm volatile("global_load_async_to_lds_b128 %0, %1, off"
               :: "v"(lds_off), "v"(ga) : "memory");
}
static __device__ inline void wait_async() {
  asm volatile("s_wait_asynccnt 0x0" ::: "memory");
}

// ---------------------------------------------------------------------------
// Staging helpers: global f32 -> fragment-linear bf16 LDS (paired-k b32 writes)
// ---------------------------------------------------------------------------
static __device__ inline void stage_a32x32(const float* __restrict__ src,
                                           long ldSrc, __bf16* dst, int tid) {
#pragma unroll
  for (int i = 0; i < 4; ++i) {
    int p = tid + i * 128;
    int m = p >> 4, kp = (p & 15) << 1;
    float f0 = src[(long)m * ldSrc + kp];
    float f1 = src[(long)m * ldSrc + kp + 1];
    int off = ((((m >> 4) * 32) + (m & 15) + (((kp >> 3) & 1) << 4)) << 4)
              + ((kp >> 4) << 3) + (kp & 7);
    *(unsigned*)&dst[off] = pack2bf(f0, f1);
  }
}

static __device__ inline void stage_b32x128(const float* __restrict__ src,
                                            long ldSrc, __bf16* dst, int tid) {
#pragma unroll
  for (int i = 0; i < 16; ++i) {
    int p = tid + i * 128;
    int n = p & 127, kp = (p >> 7) << 1;
    float f0 = src[(long)kp * ldSrc + n];
    float f1 = src[(long)(kp + 1) * ldSrc + n];
    int off = ((((n >> 4) * 32) + (n & 15) + (((kp >> 3) & 1) << 4)) << 4) + ((kp >> 4) << 3) + (kp & 7);
    *(unsigned*)&dst[off] = pack2bf(f0, f1);
  }
}

// ---------------------------------------------------------------------------
// Kernel 1: qkv = x @ Wqkv (f32->bf16 WMMA, f32 accum), RoPE fused in the
// epilogue. Block = 128 thr (4 waves), tile 32(M) x 128(N); wave (wm,wn)
// computes a 16x64 strip = 4 WMMAs/K-step sharing one A fragment.
// Double-buffered LDS: stage tile kk+32 while WMMAs consume tile kk.
// Outputs pre-swizzled into WMMA fragment-linear global layouts:
//  q: per head, [tq=pos>>4][ks=d>>5]                 A-frag (QK^T A operand)
//  k: per head, [c=pos>>6][nt=(pos>>4)&3][ks=d>>5]   B-frag (QK^T B operand)
//  v: per head, [c=pos>>6][nt=d>>4][ks=(pos>>5)&1]   B-frag (PV   B operand)
// ---------------------------------------------------------------------------
__global__ __launch_bounds__(128) void qkv_rope_kernel(
    const float* __restrict__ x, const float* __restrict__ Wqkv,
    __bf16* __restrict__ q, __bf16* __restrict__ k, __bf16* __restrict__ v) {
  __shared__ __attribute__((aligned(32))) __bf16 Asl[2][32 * 32];
  __shared__ __attribute__((aligned(32))) __bf16 Bsl[2][32 * 128];
  const int tid  = threadIdx.x;
  const int wid  = tid >> 5, lane = tid & 31;
  const int wm   = wid >> 1, wn = wid & 1;
  const int half = lane >> 4, nn = lane & 15;
  const int rowBase = blockIdx.y * 32;
  const int colBase = blockIdx.x * 128;
  const float* xA = x + (long)rowBase * DIMD;
  const float* wB = Wqkv + colBase;

  stage_a32x32(xA, DIMD, Asl[0], tid);
  stage_b32x128(wB, QKV_COLS, Bsl[0], tid);
  __syncthreads();

  v8f acc[4] = {{}, {}, {}, {}};
  for (int kk = 0; kk < DIMD; kk += 32) {
    const int cur = (kk >> 5) & 1;
    if (kk + 32 < DIMD) {                   // stage next tile into other buffer
      stage_a32x32(xA + kk + 32, DIMD, Asl[cur ^ 1], tid);
      stage_b32x128(wB + (long)(kk + 32) * QKV_COLS, QKV_COLS, Bsl[cur ^ 1], tid);
    }
    v16bf a = frag_ld(Asl[cur] + wm * 512);
#pragma unroll
    for (int nt = 0; nt < 4; ++nt) {
      v16bf b = frag_ld(Bsl[cur] + (wn * 4 + nt) * 512);
      acc[nt] = __builtin_amdgcn_wmma_f32_16x16x32_bf16(
          false, a, false, b, (short)0, acc[nt], false, false);
    }
    __syncthreads();
  }

  // Epilogue: RoPE (pair lives in adjacent lane), scatter to swizzled layouts.
#pragma unroll
  for (int nt = 0; nt < 4; ++nt) {
    const int ccol = colBase + (wn * 4 + nt) * 16 + nn;
    const int i3 = ccol >> 10;          // 0=q 1=k 2=v (uniform per 16-col tile)
    const int h  = (ccol >> 6) & 15;
    const int d  = ccol & 63;
    const float invf = __expf(-(float)(d & ~1) * 0.14391156855f); // ln(1e4)/64
#pragma unroll
    for (int j = 0; j < 8; ++j) {
      const int m   = j + 8 * half;
      const int tok = rowBase + wm * 16 + m;
      const int bi  = tok >> 11;
      const int pos = tok & (N_SEQ - 1);
      float val  = acc[nt][j];
      float outv = val;
      if (i3 < 2) {
        float pair = lane_xor_f<1>(val);
        float ang  = (float)pos * invf;
        float s, c;
        __sincosf(ang, &s, &c);
        outv = (d & 1) ? (pair * s + val * c) : (val * c - pair * s);
      }
      const long head = (long)(bi * NH + h) * HEAD_ELEMS;
      if (i3 == 0) {
        long off = head + ((((long)(pos >> 4) * 2 + (d >> 5)) * 32
                  + (pos & 15) + (((d >> 3) & 1) << 4)) << 4)
                  + (((d >> 4) & 1) << 3) + (d & 7);
        q[off] = (__bf16)outv;
      } else if (i3 == 1) {
        long off = head + ((((long)(pos >> 6) * 8 + ((pos >> 4) & 3) * 2 + (d >> 5)) * 32
                  + (pos & 15) + (((d >> 3) & 1) << 4)) << 4) + (((d >> 4) & 1) << 3) + (d & 7);
        k[off] = (__bf16)outv;
      } else {
        long off = head + ((((long)(pos >> 6) * 8 + (d >> 4) * 2 + ((pos >> 5) & 1)) * 32
                  + (d & 15) + (((pos >> 3) & 1) << 4)) << 4) + (((pos >> 4) & 1) << 3) + (pos & 7);
        v[off] = (__bf16)val;
      }
    }
  }
}

// ---------------------------------------------------------------------------
// Kernel 2: flash attention per (b,h). Block = 128 thr (4 waves), 64 Q rows.
// Q fragments load straight from global (pre-swizzled). K/V chunks are 8 KB
// contiguous (pre-swizzled): double-buffered global_load_async_to_lds_b128 —
// chunk c+1 DMA overlaps the WMMAs + softmax of chunk c (ASYNCcnt path).
// Output written in the A-fragment-linear layout kernel 3 consumes.
// ---------------------------------------------------------------------------
__global__ __launch_bounds__(128) void attention_kernel(
    const __bf16* __restrict__ q, const __bf16* __restrict__ kmat,
    const __bf16* __restrict__ vmat, __bf16* __restrict__ attn) {
  __shared__ __attribute__((aligned(32))) __bf16 Ksl[2][64 * 64];
  __shared__ __attribute__((aligned(32))) __bf16 Vsl[2][64 * 64];
  __shared__ __attribute__((aligned(32))) __bf16 Psl[4 * 16 * 64];
  const int tid  = threadIdx.x;
  const int wid  = tid >> 5, lane = tid & 31;
  const int half = lane >> 4, nn = lane & 15;
  const int bh    = blockIdx.y;
  const int qBase = blockIdx.x * 64;
  const long headOff = (long)bh * HEAD_ELEMS;
  const __bf16* qh = q    + headOff;
  const __bf16* kh = kmat + headOff;
  const __bf16* vh = vmat + headOff;

  // This wave's Q fragments (rows qBase + wid*16), direct global b128 loads.
  const int tq = (qBase >> 4) + wid;
  v16bf aQ0 = frag_ld(qh + (long)(tq * 2 + 0) * 512);
  v16bf aQ1 = frag_ld(qh + (long)(tq * 2 + 1) * 512);

  v8f o[4] = {{}, {}, {}, {}};
  float rmax[8], rsum[8];
#pragma unroll
  for (int j = 0; j < 8; ++j) { rmax[j] = -1e30f; rsum[j] = 0.0f; }

  const unsigned ksBase = (unsigned)(size_t)&Ksl[0][0];
  const unsigned vsBase = (unsigned)(size_t)&Vsl[0][0];

  // Preload chunk 0 into buffer 0.
#pragma unroll
  for (int i = 0; i < 4; ++i) {
    const unsigned boff = (unsigned)(tid + i * 128) * 16;
    async_copy_b128(ksBase + boff, (const char*)kh + boff);
    async_copy_b128(vsBase + boff, (const char*)vh + boff);
  }
  wait_async();
  __syncthreads();

  for (int c = 0; c < N_SEQ / 64; ++c) {
    const int cur = c & 1;
    if (c + 1 < N_SEQ / 64) {   // DMA next chunk into the other buffer now
      const unsigned dK = ksBase + (unsigned)(cur ^ 1) * 8192;
      const unsigned dV = vsBase + (unsigned)(cur ^ 1) * 8192;
      const char* sK = (const char*)(kh + (long)(c + 1) * 4096);
      const char* sV = (const char*)(vh + (long)(c + 1) * 4096);
#pragma unroll
      for (int i = 0; i < 4; ++i) {
        const unsigned boff = (unsigned)(tid + i * 128) * 16;
        async_copy_b128(dK + boff, sK + boff);
        async_copy_b128(dV + boff, sV + boff);
      }
    }

    // S = (Q K^T) * scale : 4 col tiles x 2 K-steps
    v8f st[4];
#pragma unroll
    for (int nt = 0; nt < 4; ++nt) {
      v8f s = {};
      v16bf b0 = frag_ld(Ksl[cur] + (nt * 2 + 0) * 512);
      s = __builtin_amdgcn_wmma_f32_16x16x32_bf16(false, aQ0, false, b0,
                                                  (short)0, s, false, false);
      v16bf b1 = frag_ld(Ksl[cur] + (nt * 2 + 1) * 512);
      s = __builtin_amdgcn_wmma_f32_16x16x32_bf16(false, aQ1, false, b1,
                                                  (short)0, s, false, false);
#pragma unroll
      for (int j = 0; j < 8; ++j) s[j] *= 0.125f;     // DH^-0.5
      st[nt] = s;
    }

    // Online softmax; row m = j+8*half lives in lanes [16*half, 16*half+16).
#pragma unroll
    for (int j = 0; j < 8; ++j) {
      float mloc = fmaxf(fmaxf(st[0][j], st[1][j]), fmaxf(st[2][j], st[3][j]));
      mloc = fmaxf(mloc, lane_xor_f<1>(mloc));
      mloc = fmaxf(mloc, lane_xor_f<2>(mloc));
      mloc = fmaxf(mloc, lane_xor_f<4>(mloc));
      mloc = fmaxf(mloc, lane_xor_f<8>(mloc));
      const float mnew  = fmaxf(rmax[j], mloc);
      const float alpha = __expf(rmax[j] - mnew);
      rmax[j] = mnew;
      float psum = 0.0f;
#pragma unroll
      for (int nt = 0; nt < 4; ++nt) {
        float p = __expf(st[nt][j] - mnew);
        st[nt][j] = p;
        psum += p;
      }
      psum += lane_xor_f<1>(psum);
      psum += lane_xor_f<2>(psum);
      psum += lane_xor_f<4>(psum);
      psum += lane_xor_f<8>(psum);
      rsum[j] = rsum[j] * alpha + psum;
      o[0][j] *= alpha; o[1][j] *= alpha; o[2][j] *= alpha; o[3][j] *= alpha;
    }

    // Restage P (16x64) into wave-private A-fragment-linear LDS.
    __bf16* pw = Psl + wid * 1024;
#pragma unroll
    for (int nt = 0; nt < 4; ++nt) {
      const int base = ((((nt >> 1) << 5) + ((nn >> 3) << 4)) << 4)
                       + ((nt & 1) << 3) + (nn & 7);
#pragma unroll
      for (int j = 0; j < 8; ++j) {
        const int m = j + 8 * half;
        pw[base + (m << 4)] = (__bf16)st[nt][j];
      }
    }
    v16bf aP0 = frag_ld(pw);
    v16bf aP1 = frag_ld(pw + 512);

    // O += P @ V
#pragma unroll
    for (int nt = 0; nt < 4; ++nt) {
      v16bf bv0 = frag_ld(Vsl[cur] + (nt * 2 + 0) * 512);
      o[nt] = __builtin_amdgcn_wmma_f32_16x16x32_bf16(false, aP0, false, bv0,
                                                      (short)0, o[nt], false, false);
      v16bf bv1 = frag_ld(Vsl[cur] + (nt * 2 + 1) * 512);
      o[nt] = __builtin_amdgcn_wmma_f32_16x16x32_bf16(false, aP1, false, bv1,
                                                      (short)0, o[nt], false, false);
    }

    wait_async();        // next chunk's DMA (overlapped above) is now done
    __syncthreads();
  }

  // Epilogue: normalize; write A-fragment-linear attn rows (token-major).
  const int b = bh >> 4, h = bh & 15;
  const long tmA = (long)(b * (N_SEQ >> 4) + tq);     // 16-row tile of token
#pragma unroll
  for (int j = 0; j < 8; ++j) {
    const int m = j + 8 * half;
    const float inv = 1.0f / rsum[j];
#pragma unroll
    for (int nt = 0; nt < 4; ++nt) {
      const long off = ((tmA * 32 + (h * 2 + (nt >> 1))) * 32
                        + m + ((nn >> 3) << 4)) * 16 + ((nt & 1) << 3) + (nn & 7);
      attn[off] = (__bf16)(o[nt][j] * inv);
    }
  }
}

// ---------------------------------------------------------------------------
// Kernel 3: out = attn @ Wout. A operands stream straight from global
// (fragment-linear bf16); B staged f32->bf16 into double-buffered LDS.
// Block = 128 thr, tile 32(M) x 128(N); 4 WMMAs/K-step per wave.
// ---------------------------------------------------------------------------
__global__ __launch_bounds__(128) void out_proj_kernel(
    const __bf16* __restrict__ attn, const float* __restrict__ Wout,
    float* __restrict__ out) {
  __shared__ __attribute__((aligned(32))) __bf16 Bsl[2][32 * 128];
  const int tid  = threadIdx.x;
  const int wid  = tid >> 5, lane = tid & 31;
  const int wm   = wid >> 1, wn = wid & 1;
  const int half = lane >> 4, nn = lane & 15;
  const int rowBase = blockIdx.y * 32;
  const int colBase = blockIdx.x * 128;
  const int tm = (rowBase >> 4) + wm;
  const float* wB = Wout + colBase;

  stage_b32x128(wB, DIMD, Bsl[0], tid);
  __syncthreads();

  v8f acc[4] = {{}, {}, {}, {}};
  for (int kb = 0; kb < DIMD / 32; ++kb) {
    const int cur = kb & 1;
    if (kb + 1 < DIMD / 32)
      stage_b32x128(wB + (long)(kb + 1) * 32 * DIMD, DIMD, Bsl[cur ^ 1], tid);
    v16bf a = frag_ld(attn + ((long)tm * 32 + kb) * 512);
#pragma unroll
    for (int nt = 0; nt < 4; ++nt) {
      v16bf b = frag_ld(Bsl[cur] + (wn * 4 + nt) * 512);
      acc[nt] = __builtin_amdgcn_wmma_f32_16x16x32_bf16(
          false, a, false, b, (short)0, acc[nt], false, false);
    }
    __syncthreads();
  }
#pragma unroll
  for (int nt = 0; nt < 4; ++nt)
#pragma unroll
    for (int j = 0; j < 8; ++j) {
      const int m = j + 8 * half;
      out[(long)(rowBase + wm * 16 + m) * DIMD + colBase + (wn * 4 + nt) * 16 + nn]
          = acc[nt][j];
    }
}

// ---------------------------------------------------------------------------
extern "C" void kernel_launch(void* const* d_in, const int* in_sizes, int n_in,
                              void* d_out, int out_size, void* d_ws, size_t ws_size,
                              hipStream_t stream) {
  (void)in_sizes; (void)n_in; (void)out_size; (void)ws_size;
  const float* x    = (const float*)d_in[0];   // [B,N,DIM] f32
  const float* Wqkv = (const float*)d_in[1];   // [DIM,3*H*DH] f32
  const float* Wout = (const float*)d_in[2];   // [H*DH,DIM] f32
  float* out = (float*)d_out;                  // [B,N,DIM] f32

  // Workspace: q,k,v (fragment-linear per head) + attn (fragment-linear rows)
  // 4 x 8 MB = 32 MB bf16.
  __bf16* q    = (__bf16*)d_ws;
  __bf16* k    = q + TENSOR_ELEMS;
  __bf16* v    = k + TENSOR_ELEMS;
  __bf16* attn = v + TENSOR_ELEMS;

  qkv_rope_kernel<<<dim3(QKV_COLS / 128, M_ROWS / 32), 128, 0, stream>>>(
      x, Wqkv, q, k, v);
  attention_kernel<<<dim3(N_SEQ / 64, B_SZ * NH), 128, 0, stream>>>(
      q, k, v, attn);
  out_proj_kernel<<<dim3(DIMD / 128, M_ROWS / 32), 128, 0, stream>>>(
      attn, Wout, out);
}
